// DecoderLayer_84817014161625
// MI455X (gfx1250) — compile-verified, matrix operand order checked by reading, not run on hardware
//
#include <hip/hip_runtime.h>
#include <math.h>

typedef __attribute__((ext_vector_type(16))) _Float16 v16h;
typedef __attribute__((ext_vector_type(8)))  float    v8f;

#define H_DIM   1024
#define S_LEN   1024
#define NHEAD   16
#define HEAD_D  64
#define B_SZ    4
#define INNER_D 4096
#define LN_EPS  1e-5f

union Frag16 {
  v16h v;
  _Float16 h[16];
  unsigned u[8];
};

// K-index of the j-th half-pair for a 16-bit A/B fragment (ISA 7.12.2 layout):
// lanes 0-15 (hi=0): VGPR0-3 hold K=0..7, VGPR4-7 hold K=16..23
// lanes 16-31 (hi=1): VGPR0-3 hold K=8..15, VGPR4-7 hold K=24..31
__device__ __forceinline__ int pair_k(int j, int hi) {
  return (j < 4) ? (hi * 8 + 2 * j) : (16 + hi * 8 + 2 * (j - 4));
}

// Load a 16x32 (MxK) f16 fragment from LDS (row-major halves, 4B-aligned pairs;
// each half of the fragment is 8 contiguous halves -> two ds_load_b128).
__device__ __forceinline__ v16h load_frag_lds(const _Float16* smem, int row0,
                                              int ld, int kbase) {
  int lane = threadIdx.x & 31;
  int m = lane & 15, hi = lane >> 4;
  const _Float16* p = smem + (size_t)(row0 + m) * ld + kbase;
  Frag16 f;
#pragma unroll
  for (int j = 0; j < 8; ++j)
    f.u[j] = *(const unsigned*)(p + pair_k(j, hi));
  return f.v;
}

// Load a 16x32 fragment directly from global fp32 (row-major), converting to f16.
__device__ __forceinline__ v16h load_frag_g32(const float* base, int ldg,
                                              int kbase) {
  int lane = threadIdx.x & 31;
  int m = lane & 15, hi = lane >> 4;
  const float* p = base + (size_t)m * ldg + kbase;
  Frag16 f;
#pragma unroll
  for (int j = 0; j < 8; ++j) {
    int k = pair_k(j, hi);
    f.h[2 * j]     = (_Float16)p[k];
    f.h[2 * j + 1] = (_Float16)p[k + 1];
  }
  return f.v;
}

__device__ __forceinline__ v8f wmma16(v16h a, v16h b, v8f c) {
  return __builtin_amdgcn_wmma_f32_16x16x32_f16(false, a, false, b, (short)0,
                                                c, false, false);
}

// Store a 16x16 f32 C fragment (+bias, optional relu) to global row-major C.
__device__ __forceinline__ void store_frag(v8f c, float* __restrict__ C, int N,
                                           int row0, int col0,
                                           const float* __restrict__ bias,
                                           int relu) {
  int lane = threadIdx.x & 31;
  int hi = lane >> 4, n = lane & 15;
  float bv = bias[col0 + n];
#pragma unroll
  for (int r = 0; r < 8; ++r) {
    float v = c[r] + bv;
    if (relu) v = fmaxf(v, 0.f);
    C[(size_t)(row0 + hi * 8 + r) * N + col0 + n] = v;
  }
}

// ---------------------------------------------------------------------------
// GEMM: C[M,N] = A[M,K] * W[N,K]^T + bias, optional ReLU.
// Block 256 threads (8 waves, 4x2), block tile 128x128, wave tile 32x64,
// K chunk 32. Register-stage prefetch: chunk i+1 global loads are issued
// right after the barrier and overlap with chunk i's WMMAs.
// ---------------------------------------------------------------------------
__global__ __launch_bounds__(256) void gemm_bias_kernel(
    const float* __restrict__ A, const float* __restrict__ W,
    const float* __restrict__ bias, float* __restrict__ C, int M, int N, int K,
    int relu) {
  __shared__ _Float16 As[128 * 40];
  __shared__ _Float16 Bs[128 * 40];
  const int tid = threadIdx.x;
  const int wid = tid >> 5;
  const int wm = (wid & 3) * 32;   // 4 waves along M
  const int wn = (wid >> 2) * 64;  // 2 waves along N
  const int bm = blockIdx.y * 128;
  const int bn = blockIdx.x * 128;

  v8f acc[2][4];
#pragma unroll
  for (int mi = 0; mi < 2; ++mi)
#pragma unroll
    for (int ni = 0; ni < 4; ++ni) acc[mi][ni] = {};

  // each thread stages 16 floats of A and 16 of B per chunk
  const int row = tid >> 1, cb = (tid & 1) * 16;
  const float* aptr = A + (size_t)(bm + row) * K + cb;
  const float* bptr = W + (size_t)(bn + row) * K + cb;

  float4 ra[4], rb[4];
#pragma unroll
  for (int j = 0; j < 4; ++j) {
    ra[j] = *(const float4*)(aptr + 4 * j);
    rb[j] = *(const float4*)(bptr + 4 * j);
  }

  for (int kk = 0; kk < K; kk += 32) {
    // commit staged registers to LDS (f32 -> f16)
    {
      _Float16* da = &As[row * 40 + cb];
      _Float16* db = &Bs[row * 40 + cb];
#pragma unroll
      for (int j = 0; j < 4; ++j) {
        da[4 * j + 0] = (_Float16)ra[j].x; da[4 * j + 1] = (_Float16)ra[j].y;
        da[4 * j + 2] = (_Float16)ra[j].z; da[4 * j + 3] = (_Float16)ra[j].w;
        db[4 * j + 0] = (_Float16)rb[j].x; db[4 * j + 1] = (_Float16)rb[j].y;
        db[4 * j + 2] = (_Float16)rb[j].z; db[4 * j + 3] = (_Float16)rb[j].w;
      }
    }
    __syncthreads();

    // prefetch next K chunk while this chunk's WMMAs run
    if (kk + 32 < K) {
      const float* an = aptr + kk + 32;
      const float* bn_ = bptr + kk + 32;
#pragma unroll
      for (int j = 0; j < 4; ++j) {
        ra[j] = *(const float4*)(an + 4 * j);
        rb[j] = *(const float4*)(bn_ + 4 * j);
      }
    }

    v16h af[2], bf[4];
#pragma unroll
    for (int mi = 0; mi < 2; ++mi)
      af[mi] = load_frag_lds(As, wm + 16 * mi, 40, 0);
#pragma unroll
    for (int ni = 0; ni < 4; ++ni)
      bf[ni] = load_frag_lds(Bs, wn + 16 * ni, 40, 0);
#pragma unroll
    for (int mi = 0; mi < 2; ++mi)
#pragma unroll
      for (int ni = 0; ni < 4; ++ni)
        acc[mi][ni] = wmma16(af[mi], bf[ni], acc[mi][ni]);

    __syncthreads();
  }

#pragma unroll
  for (int mi = 0; mi < 2; ++mi)
#pragma unroll
    for (int ni = 0; ni < 4; ++ni)
      store_frag(acc[mi][ni], C, N, bm + wm + 16 * mi, bn + wn + 16 * ni, bias,
                 relu);
}

// ---------------------------------------------------------------------------
// Flash attention. Q,K,V,O are [B,S,H] with head h occupying cols h*64..h*64+63.
// Grid (S/64, NH, B); 128 threads = 4 waves; wave owns 16 query rows.
// Online softmax over 32-key chunks; scores & PV via WMMA.
// ---------------------------------------------------------------------------
__global__ __launch_bounds__(128) void attn_kernel(
    const float* __restrict__ Q, const float* __restrict__ Km,
    const float* __restrict__ V, float* __restrict__ O) {
  __shared__ _Float16 Ks[32 * 72];      // keys x d
  __shared__ _Float16 Vt[64 * 40];      // d x keys (transposed)
  __shared__ _Float16 Ps[4][16 * 40];   // per-wave P relayout buffer

  const int tid = threadIdx.x;
  const int lane = tid & 31;
  const int wid = tid >> 5;
  const int h = blockIdx.y;
  const int b = blockIdx.z;
  const int q0 = blockIdx.x * 64 + wid * 16;

  const size_t headoff = (size_t)h * HEAD_D;
  const float* qbase = Q + ((size_t)b * S_LEN + q0) * H_DIM + headoff;
  v16h qa0 = load_frag_g32(qbase, H_DIM, 0);
  v16h qa1 = load_frag_g32(qbase, H_DIM, 32);

  v8f o0 = {}, o1 = {}, o2 = {}, o3 = {};
  float mrow[8], lrow[8];
#pragma unroll
  for (int r = 0; r < 8; ++r) { mrow[r] = -3.0e38f; lrow[r] = 0.f; }

  const float scale = 0.125f;            // HD^-0.5 = 1/8
  const int key = tid >> 2;              // 0..31
  const int db = (tid & 3) * 16;         // 0,16,32,48

  for (int kc = 0; kc < S_LEN; kc += 32) {
    __syncthreads();
    {
      const size_t roff = ((size_t)b * S_LEN + kc + key) * H_DIM + headoff + db;
      const float* kr = Km + roff;
      const float* vr = V + roff;
#pragma unroll
      for (int j = 0; j < 16; j += 4) {
        float4 t = *(const float4*)(kr + j);
        _Float16* d = &Ks[key * 72 + db + j];
        d[0] = (_Float16)t.x; d[1] = (_Float16)t.y;
        d[2] = (_Float16)t.z; d[3] = (_Float16)t.w;
        float4 u = *(const float4*)(vr + j);
        Vt[(db + j + 0) * 40 + key] = (_Float16)u.x;
        Vt[(db + j + 1) * 40 + key] = (_Float16)u.y;
        Vt[(db + j + 2) * 40 + key] = (_Float16)u.z;
        Vt[(db + j + 3) * 40 + key] = (_Float16)u.w;
      }
    }
    __syncthreads();

    // scores: two 16-key groups, K-dim = HD = 64 -> two k32 steps each
    v8f s0 = {}, s1 = {};
    s0 = wmma16(qa0, load_frag_lds(Ks, 0, 72, 0), s0);
    s0 = wmma16(qa1, load_frag_lds(Ks, 0, 72, 32), s0);
    s1 = wmma16(qa0, load_frag_lds(Ks, 16, 72, 0), s1);
    s1 = wmma16(qa1, load_frag_lds(Ks, 16, 72, 32), s1);

    // online softmax (rows live in VGPR index r, cols in lanes 0-15 per half)
    float alpha[8];
#pragma unroll
    for (int r = 0; r < 8; ++r) {
      float x0 = s0[r] * scale, x1 = s1[r] * scale;
      float rm = fmaxf(x0, x1);
#pragma unroll
      for (int off = 1; off < 16; off <<= 1)
        rm = fmaxf(rm, __shfl_xor(rm, off, 32));
      float nm = fmaxf(mrow[r], rm);
      float a = __expf(mrow[r] - nm);
      float p0 = __expf(x0 - nm);
      float p1 = __expf(x1 - nm);
      float rs = p0 + p1;
#pragma unroll
      for (int off = 1; off < 16; off <<= 1)
        rs += __shfl_xor(rs, off, 32);
      mrow[r] = nm;
      lrow[r] = lrow[r] * a + rs;
      alpha[r] = a;
      s0[r] = p0; s1[r] = p1;
    }
#pragma unroll
    for (int r = 0; r < 8; ++r) {
      o0[r] *= alpha[r]; o1[r] *= alpha[r];
      o2[r] *= alpha[r]; o3[r] *= alpha[r];
    }

    // relayout P (C-layout) -> A-fragment via wave-private LDS
    {
      int hi = lane >> 4, n = lane & 15;
      _Float16* pp = &Ps[wid][0];
#pragma unroll
      for (int r = 0; r < 8; ++r) {
        pp[(hi * 8 + r) * 40 + n]      = (_Float16)s0[r];
        pp[(hi * 8 + r) * 40 + 16 + n] = (_Float16)s1[r];
      }
    }
    v16h pa = load_frag_lds(&Ps[wid][0], 0, 40, 0);  // 16 q x 32 keys
    o0 = wmma16(pa, load_frag_lds(Vt, 0, 40, 0), o0);
    o1 = wmma16(pa, load_frag_lds(Vt, 16, 40, 0), o1);
    o2 = wmma16(pa, load_frag_lds(Vt, 32, 40, 0), o2);
    o3 = wmma16(pa, load_frag_lds(Vt, 48, 40, 0), o3);
  }

  // epilogue: normalize by l, store in (b, s, h*64 + d) layout
  {
    int hi = lane >> 4, n = lane & 15;
    float* obase = O + ((size_t)b * S_LEN + q0) * H_DIM + headoff;
#pragma unroll
    for (int r = 0; r < 8; ++r) {
      float inv = 1.0f / lrow[r];
      float* row = obase + (size_t)(hi * 8 + r) * H_DIM;
      row[n]      = o0[r] * inv;
      row[16 + n] = o1[r] * inv;
      row[32 + n] = o2[r] * inv;
      row[48 + n] = o3[r] * inv;
    }
  }
}

// ---------------------------------------------------------------------------
// out = LayerNorm(a + res) * g + b  (one H=1024 row per 256-thread block)
// Safe in-place on `res` (all reads before any write within the block's row).
// ---------------------------------------------------------------------------
__global__ __launch_bounds__(256) void ln_kernel(
    const float* __restrict__ a, const float* __restrict__ res,
    const float* __restrict__ g, const float* __restrict__ be,
    float* __restrict__ out) {
  __shared__ float sred[16];
  const int tid = threadIdx.x;
  const size_t off = (size_t)blockIdx.x * H_DIM + tid * 4;
  float4 av = *(const float4*)(a + off);
  float4 rv = *(const float4*)(res + off);
  float x0 = av.x + rv.x, x1 = av.y + rv.y, x2 = av.z + rv.z, x3 = av.w + rv.w;
  float s = x0 + x1 + x2 + x3;
  float s2 = x0 * x0 + x1 * x1 + x2 * x2 + x3 * x3;
#pragma unroll
  for (int m = 16; m >= 1; m >>= 1) {
    s += __shfl_xor(s, m, 32);
    s2 += __shfl_xor(s2, m, 32);
  }
  int lane = tid & 31, wid = tid >> 5;
  if (lane == 0) { sred[wid] = s; sred[8 + wid] = s2; }
  __syncthreads();
  if (wid == 0) {
    float ts = (lane < 8) ? sred[lane] : 0.f;
    float ts2 = (lane < 8) ? sred[8 + lane] : 0.f;
#pragma unroll
    for (int m = 4; m >= 1; m >>= 1) {
      ts += __shfl_xor(ts, m, 32);
      ts2 += __shfl_xor(ts2, m, 32);
    }
    if (lane == 0) { sred[0] = ts; sred[1] = ts2; }
  }
  __syncthreads();
  float mean = sred[0] * (1.0f / H_DIM);
  float var = sred[1] * (1.0f / H_DIM) - mean * mean;
  float rstd = rsqrtf(var + LN_EPS);
  const int c = tid * 4;
  float4 gv = *(const float4*)(g + c);
  float4 bv = *(const float4*)(be + c);
  float4 ov;
  ov.x = (x0 - mean) * rstd * gv.x + bv.x;
  ov.y = (x1 - mean) * rstd * gv.y + bv.y;
  ov.z = (x2 - mean) * rstd * gv.z + bv.z;
  ov.w = (x3 - mean) * rstd * gv.w + bv.w;
  *(float4*)(out + off) = ov;
}

// ---------------------------------------------------------------------------
// d_in layout (setup_inputs dict order, params flattened in insertion order):
// 0 input, 1 enc_output, 2 paddingMask, 3 lookaheadMask,
// 4-11 attn1 {q.w,q.b,k.w,k.b,v.w,v.b,o.w,o.b}, 12-13 ln1 {g,b},
// 14-21 attn2 {...}, 22-23 ln2, 24-27 ffn {fc1.w,fc1.b,fc2.w,fc2.b}, 28-29 ln3
// ---------------------------------------------------------------------------
extern "C" void kernel_launch(void* const* d_in, const int* in_sizes, int n_in,
                              void* d_out, int out_size, void* d_ws,
                              size_t ws_size, hipStream_t stream) {
  (void)in_sizes; (void)n_in; (void)out_size; (void)ws_size;
  const float* x_in = (const float*)d_in[0];
  const float* enc = (const float*)d_in[1];

  const size_t NT = (size_t)B_SZ * S_LEN * H_DIM;  // 4M floats / 16 MB
  float* ws = (float*)d_ws;
  float* bufQ = ws;            // q
  float* bufK = ws + NT;       // k
  float* bufV = ws + 2 * NT;   // v
  float* bufA = ws + 3 * NT;   // attention output (pre O-proj)
  float* bufE = ws + 4 * NT;   // projection / ffn output
  float* bufX = ws + 5 * NT;   // residual stream
  float* bufF = ws;            // ffn inner (64 MB) overlaps Q..A (dead then)

  const int M = B_SZ * S_LEN;  // 4096
  auto gemm = [&](const float* A, int wi, int bi, float* C, int N, int K,
                  int relu) {
    dim3 grid(N / 128, M / 128);
    gemm_bias_kernel<<<grid, 256, 0, stream>>>(
        A, (const float*)d_in[wi], (const float*)d_in[bi], C, M, N, K, relu);
  };
  dim3 agrid(S_LEN / 64, NHEAD, B_SZ);

  // --- self-attention (masks are ignored, matching the reference) ---
  gemm(x_in, 4, 5, bufQ, H_DIM, H_DIM, 0);
  gemm(x_in, 6, 7, bufK, H_DIM, H_DIM, 0);
  gemm(x_in, 8, 9, bufV, H_DIM, H_DIM, 0);
  attn_kernel<<<agrid, 128, 0, stream>>>(bufQ, bufK, bufV, bufA);
  gemm(bufA, 10, 11, bufE, H_DIM, H_DIM, 0);
  ln_kernel<<<M, 256, 0, stream>>>(bufE, x_in, (const float*)d_in[12],
                                   (const float*)d_in[13], bufX);

  // --- cross-attention with encoder output ---
  gemm(bufX, 14, 15, bufQ, H_DIM, H_DIM, 0);
  gemm(enc, 16, 17, bufK, H_DIM, H_DIM, 0);
  gemm(enc, 18, 19, bufV, H_DIM, H_DIM, 0);
  attn_kernel<<<agrid, 128, 0, stream>>>(bufQ, bufK, bufV, bufA);
  gemm(bufA, 20, 21, bufE, H_DIM, H_DIM, 0);
  ln_kernel<<<M, 256, 0, stream>>>(bufE, bufX, (const float*)d_in[22],
                                   (const float*)d_in[23], bufX);

  // --- FFN ---
  gemm(bufX, 24, 25, bufF, INNER_D, H_DIM, 1);   // relu fused
  gemm(bufF, 26, 27, bufE, H_DIM, INNER_D, 0);
  ln_kernel<<<M, 256, 0, stream>>>(bufE, bufX, (const float*)d_in[28],
                                   (const float*)d_in[29], (float*)d_out);
}
